// InverseFoldingGraghInputEmbedder_28845000360675
// MI455X (gfx1250) — matrix operand order repaired and checked by reading, not compiled, not run on hardware
//
#include <hip/hip_runtime.h>
#include <hip/hip_bf16.h>

typedef _Float16 half_t;
typedef __attribute__((ext_vector_type(16))) _Float16 v16h;
typedef __attribute__((ext_vector_type(8)))  _Float16 v8h;
typedef __attribute__((ext_vector_type(8)))  float    v8f;
typedef __attribute__((ext_vector_type(4)))  float    f4;

#define B_   2
#define N_   768
#define TOPK 30
#define DP   128      // d_pair
#define DM   256      // d_msa
#define NBIN 65       // 2*32+1
#define INFV 1e10f
#define BIGV 3e38f

// output layout (floats): s | ang | dst | relpos
#define OFF_S    0
#define OFF_ANG  (B_*N_*DM)                         // 393216
#define OFF_DST  (OFF_ANG + B_*N_*TOPK*DP)          // 6291456
#define OFF_RP   (OFF_DST + B_*N_*TOPK*DP)          // 12189696

// ws layout (bytes): packed f16 Wa2 | packed f16 Wd2 | relpos table f32
#define WS_W2A 0
#define WS_W2D 32768
#define WS_TAB 65536

__device__ __forceinline__ float gelu_tanh(float x) {
    float x3 = x * x * x;
    return 0.5f * x * (1.0f + tanhf(0.7978845608028654f * (x + 0.044715f * x3)));
}

// ---------------------------------------------------------------------------
// prep: pack W2 weights into f16 WMMA-B-fragment order, build relpos table.
// packed[((g*128 + n)*16 + q)] = W2[(g*16+q)*128 + n], g = kt*2 + khalf (0..7)
// so a lane's 16 B-frag halves for (kt, khalf, n) are contiguous (32 bytes).
// ---------------------------------------------------------------------------
__global__ void prep_kernel(const float* __restrict__ Wa2,
                            const float* __restrict__ Wd2,
                            const float* __restrict__ Wrp,
                            const float* __restrict__ brp,
                            half_t* __restrict__ w2a_pk,
                            half_t* __restrict__ w2d_pk,
                            float* __restrict__ table) {
    int stride = gridDim.x * blockDim.x;
    for (int idx = blockIdx.x * blockDim.x + threadIdx.x; idx < DP * DP; idx += stride) {
        int q = idx & 15;
        int n = (idx >> 4) & 127;
        int g = idx >> 11;
        int k = g * 16 + q;
        w2a_pk[idx] = (half_t)Wa2[k * DP + n];
        w2d_pk[idx] = (half_t)Wd2[k * DP + n];
    }
    for (int idx = blockIdx.x * blockDim.x + threadIdx.x; idx < NBIN * DP; idx += stride)
        table[idx] = Wrp[idx] + brp[idx & 127];
}

// ---------------------------------------------------------------------------
// s = ones @ W_tf + b_tf  ->  broadcast W_tf[c]+b_tf[c] over B*N rows
// ---------------------------------------------------------------------------
__global__ void s_fill_kernel(const float* __restrict__ W_tf,
                              const float* __restrict__ b_tf,
                              float* __restrict__ out_s) {
    int stride = gridDim.x * blockDim.x;
    for (int idx = blockIdx.x * blockDim.x + threadIdx.x; idx < B_ * N_ * DM; idx += stride)
        out_s[idx] = W_tf[idx & (DM - 1)] + b_tf[idx & (DM - 1)];
}

// ---------------------------------------------------------------------------
// fused: distances -> top-30 -> geometric features -> 2x MLP embed (WMMA).
// one block per (b,i) residue; 128 threads = 4 waves.
// ---------------------------------------------------------------------------
__global__ __launch_bounds__(128)
void pair_embed_kernel(const int*   __restrict__ ridx,
                       const float* __restrict__ frames,   // (B,N,4,4)
                       const float* __restrict__ smask,
                       const float* __restrict__ Wa1, const float* __restrict__ ba1,
                       const float* __restrict__ ba2,
                       const float* __restrict__ Wd1, const float* __restrict__ bd1,
                       const float* __restrict__ bd2,
                       const half_t* __restrict__ w2a_pk,
                       const half_t* __restrict__ w2d_pk,
                       float* __restrict__ out_ang,
                       float* __restrict__ out_dst) {
    __shared__ __attribute__((aligned(16))) float  s_d4[N_];
    __shared__ float  s_rv[128];
    __shared__ int    s_ri[128];
    __shared__ int    s_idx[32];
    __shared__ float  s_fa[32][8];   // [eye, conn, dir0, dir1, dir2]
    __shared__ float  s_fd[32][4];   // [eye, conn, dist]
    __shared__ __attribute__((aligned(16))) half_t s_h[2][32][DP];  // hidden (f16)

    const int tid  = threadIdx.x;
    const int row  = blockIdx.x;        // b*N + i
    const int b    = row / N_;
    const int i    = row - b * N_;

    // local frame of residue i
    const float* fi = frames + (size_t)row * 16;
    float R[3][3], ti[3];
    #pragma unroll
    for (int x = 0; x < 3; ++x) {
        #pragma unroll
        for (int y = 0; y < 3; ++y) R[x][y] = fi[x * 4 + y];
        ti[x] = fi[x * 4 + 3];
    }
    const int   ri_i = ridx[row];
    const float sm_i = smask[row];

    // ---- masked distance row ----
    for (int j = tid; j < N_; j += 128) {
        const float* fj = frames + (size_t)(b * N_ + j) * 16;
        float dx = fj[3] - ti[0], dy = fj[7] - ti[1], dz = fj[11] - ti[2];
        float d = sqrtf(dx * dx + dy * dy + dz * dz);
        float v = d;
        if (j == i) v = INFV;
        if (!(sm_i != 0.0f && smask[b * N_ + j] != 0.0f)) v = INFV;
        int rd = ri_i - ridx[b * N_ + j];
        if (rd == 1 || rd == -1) v = -INFV;   // connect wins over mask
        s_d4[j] = v;
    }
    __syncthreads();

    // ---- iterative argmin top-30 (tie -> lower index, matches top_k) ----
    for (int k = 0; k < TOPK; ++k) {
        float bv = BIGV; int bi = N_;
        for (int j = tid; j < N_; j += 128) {
            float v = s_d4[j];
            if (v < bv || (v == bv && j < bi)) { bv = v; bi = j; }
        }
        s_rv[tid] = bv; s_ri[tid] = bi;
        __syncthreads();
        for (int s = 64; s > 0; s >>= 1) {
            if (tid < s) {
                float v2 = s_rv[tid + s]; int i2 = s_ri[tid + s];
                if (v2 < s_rv[tid] || (v2 == s_rv[tid] && i2 < s_ri[tid])) {
                    s_rv[tid] = v2; s_ri[tid] = i2;
                }
            }
            __syncthreads();
        }
        if (tid == 0) { s_idx[k] = s_ri[0]; s_d4[s_ri[0]] = BIGV; }
        __syncthreads();
    }

    // ---- geometric features for the 30 neighbors (rows 30/31 = pad) ----
    if (tid < 32) {
        if (tid < TOPK) {
            int j = s_idx[tid];
            const float* fj = frames + (size_t)(b * N_ + j) * 16;
            float v0 = fj[3] - ti[0], v1 = fj[7] - ti[1], v2 = fj[11] - ti[2];
            float d = sqrtf(v0 * v0 + v1 * v1 + v2 * v2);
            float tr0 = R[0][0] * v0 + R[1][0] * v1 + R[2][0] * v2;   // R^T v
            float tr1 = R[0][1] * v0 + R[1][1] * v1 + R[2][1] * v2;
            float tr2 = R[0][2] * v0 + R[1][2] * v1 + R[2][2] * v2;
            float inv = (d > 0.0f) ? 1.0f / d : 0.0f;
            float eye  = (j == i) ? 1.0f : 0.0f;
            int   rd   = ri_i - ridx[b * N_ + j];
            float conn = (rd == 1 || rd == -1) ? 1.0f : 0.0f;
            s_fa[tid][0] = eye;  s_fa[tid][1] = conn;
            s_fa[tid][2] = tr0 * inv; s_fa[tid][3] = tr1 * inv; s_fa[tid][4] = tr2 * inv;
            s_fd[tid][0] = eye;  s_fd[tid][1] = conn; s_fd[tid][2] = d;
        } else {
            #pragma unroll
            for (int f = 0; f < 5; ++f) s_fa[tid][f] = 0.0f;
            #pragma unroll
            for (int f = 0; f < 3; ++f) s_fd[tid][f] = 0.0f;
        }
    }
    __syncthreads();

    // ---- layer 1 + gelu -> f16 hidden in LDS ----
    for (int idx = tid; idx < 32 * DP; idx += 128) {
        int r = idx >> 7, c = idx & 127;
        float acc = ba1[c];
        #pragma unroll
        for (int f = 0; f < 5; ++f) acc += s_fa[r][f] * Wa1[f * DP + c];
        s_h[0][r][c] = (half_t)gelu_tanh(acc);
        float accd = bd1[c];
        #pragma unroll
        for (int f = 0; f < 3; ++f) accd += s_fd[r][f] * Wd1[f * DP + c];
        s_h[1][r][c] = (half_t)gelu_tanh(accd);
    }
    __syncthreads();

    // ---- layer 2: (32x128)@(128x128) via v_wmma_f32_16x16x32_f16 ----
    const int wave  = tid >> 5;
    const int lane  = tid & 31;
    const int mrow  = lane & 15;
    const int khalf = lane >> 4;                 // 0: lanes 0-15, 1: lanes 16-31
    const size_t rowbase = (size_t)row * TOPK;

    #pragma unroll
    for (int e = 0; e < 2; ++e) {
        const half_t* w2pk = e ? w2d_pk : w2a_pk;
        const float*  b2   = e ? bd2 : ba2;
        float*        dst  = e ? out_dst : out_ang;
        #pragma unroll
        for (int ntl = 0; ntl < 2; ++ntl) {
            const int n0 = wave * 32 + ntl * 16;
            const int n  = n0 + mrow;
            v8f acc0 = {}, acc1 = {};
            #pragma unroll
            for (int kt = 0; kt < 4; ++kt) {
                const int k0 = kt * 32;
                // A fragments (contiguous K, matching packed B): lanes<16 hold K {k0..k0+15}
                v8h a0lo = *(const v8h*)&s_h[e][mrow][k0 + khalf * 16];
                v8h a0hi = *(const v8h*)&s_h[e][mrow][k0 + khalf * 16 + 8];
                v8h a1lo = *(const v8h*)&s_h[e][16 + mrow][k0 + khalf * 16];
                v8h a1hi = *(const v8h*)&s_h[e][16 + mrow][k0 + khalf * 16 + 8];
                v16h a0, a1;
                #pragma unroll
                for (int q = 0; q < 8; ++q) {
                    a0[q] = a0lo[q]; a0[8 + q] = a0hi[q];
                    a1[q] = a1lo[q]; a1[8 + q] = a1hi[q];
                }
                // B fragment: pre-packed contiguous 16 halves per (kt,khalf,n)
                const int g = kt * 2 + khalf;
                const v8h* bp = (const v8h*)(w2pk + ((size_t)(g * DP + n) << 4));
                v8h blo = bp[0], bhi = bp[1];
                v16h bf;
                #pragma unroll
                for (int q = 0; q < 8; ++q) { bf[q] = blo[q]; bf[8 + q] = bhi[q]; }

                acc0 = __builtin_amdgcn_wmma_f32_16x16x32_f16(false, a0, false, bf,
                                                              (short)0, acc0, false, false);
                acc1 = __builtin_amdgcn_wmma_f32_16x16x32_f16(false, a1, false, bf,
                                                              (short)0, acc1, false, false);
            }
            // C/D layout: VGPR r -> M = r + 8*khalf; N = lane&15
            const float bias = b2[n];
            #pragma unroll
            for (int r = 0; r < 8; ++r) {
                int m = r + 8 * khalf;            // tile 0: rows 0..15
                dst[(rowbase + m) * DP + n] = acc0[r] + bias;
                int m1 = 16 + m;                  // tile 1: rows 16..31 (drop >=30)
                if (m1 < TOPK)
                    dst[(rowbase + m1) * DP + n] = acc1[r] + bias;
            }
        }
    }
}

// ---------------------------------------------------------------------------
// relpos: 604MB write-only stream -> table gather + non-temporal b128 stores
// ---------------------------------------------------------------------------
__global__ __launch_bounds__(256)
void relpos_kernel(const int* __restrict__ ridx,
                   const float* __restrict__ table,
                   float* __restrict__ out) {
    const size_t total4 = (size_t)B_ * N_ * N_ * (DP / 4);   // 37,748,736
    size_t tid = (size_t)blockIdx.x * blockDim.x + threadIdx.x;
    if (tid >= total4) return;
    int    c4   = (int)(tid & 31);
    size_t rest = tid >> 5;
    int j = (int)(rest % N_); rest /= N_;
    int i = (int)(rest % N_);
    int b = (int)(rest / N_);
    int d = ridx[b * N_ + i] - ridx[b * N_ + j];
    d = (d < -32 ? -32 : (d > 32 ? 32 : d)) + 32;
    f4 v = ((const f4*)(table + d * DP))[c4];
    __builtin_nontemporal_store(v, (f4*)out + tid);
}

extern "C" void kernel_launch(void* const* d_in, const int* in_sizes, int n_in,
                              void* d_out, int out_size, void* d_ws, size_t ws_size,
                              hipStream_t stream) {
    const int*   ridx   = (const int*)d_in[1];
    const float* frames = (const float*)d_in[2];
    const float* smask  = (const float*)d_in[3];
    const float* W_tf   = (const float*)d_in[4];
    const float* b_tf   = (const float*)d_in[5];
    const float* Wa1    = (const float*)d_in[6];
    const float* ba1    = (const float*)d_in[7];
    const float* Wa2    = (const float*)d_in[8];
    const float* ba2    = (const float*)d_in[9];
    const float* Wd1    = (const float*)d_in[10];
    const float* bd1    = (const float*)d_in[11];
    const float* Wd2    = (const float*)d_in[12];
    const float* bd2    = (const float*)d_in[13];
    const float* W_rp   = (const float*)d_in[14];
    const float* b_rp   = (const float*)d_in[15];

    float* out = (float*)d_out;
    half_t* w2a_pk = (half_t*)((char*)d_ws + WS_W2A);
    half_t* w2d_pk = (half_t*)((char*)d_ws + WS_W2D);
    float*  table  = (float*)((char*)d_ws + WS_TAB);

    prep_kernel<<<64, 256, 0, stream>>>(Wa2, Wd2, W_rp, b_rp, w2a_pk, w2d_pk, table);

    s_fill_kernel<<<384, 256, 0, stream>>>(W_tf, b_tf, out + OFF_S);

    pair_embed_kernel<<<B_ * N_, 128, 0, stream>>>(
        ridx, frames, smask, Wa1, ba1, ba2, Wd1, bd1, bd2,
        w2a_pk, w2d_pk, out + OFF_ANG, out + OFF_DST);

    const size_t total4 = (size_t)B_ * N_ * N_ * (DP / 4);
    relpos_kernel<<<(unsigned)((total4 + 255) / 256), 256, 0, stream>>>(
        ridx, table, out + OFF_RP);
}
